// PointNetSetAbstraction_33552284516958
// MI455X (gfx1250) — matrix operand order repaired and checked by reading, not compiled, not run on hardware
//
#include <hip/hip_runtime.h>

#define DEV __device__ __forceinline__

typedef _Float16 v8h  __attribute__((ext_vector_type(8)));
typedef _Float16 v16h __attribute__((ext_vector_type(16)));
typedef float    v8f  __attribute__((ext_vector_type(8)));

// ---------------- problem constants ----------------
constexpr int   B   = 16;
constexpr int   N   = 4096;
constexpr int   D   = 64;
constexpr int   S   = 1024;   // NPOINT
constexpr int   K   = 32;     // NSAMPLE
constexpr int   C0  = 67;     // 3 + D input channels
constexpr int   C0P = 96;     // padded to 3 K-tiles of 32
constexpr int   C1  = 64;
constexpr int   C2  = 64;
constexpr int   C3  = 128;
constexpr int   M   = B * S * K;          // 524288 grouped rows
constexpr float R2  = 0.2f * 0.2f;
constexpr float EPSV = 1e-5f;

// ---------------- workspace layout (bytes) ----------------
constexpr size_t OFF_NEWXYZ = 0;                                   // B*S*3 f32
constexpr size_t OFF_XYZT   = OFF_NEWXYZ + (size_t)B * S * 3 * 4;  // B*N*3 f32
constexpr size_t OFF_PTST   = OFF_XYZT   + (size_t)B * N * 3 * 4;  // B*N*D f16
constexpr size_t OFF_W0H    = OFF_PTST   + (size_t)B * N * D * 2;  // C1*C0P f16
constexpr size_t OFF_W1H    = OFF_W0H    + (size_t)C1 * C0P * 2;   // C2*C1 f16
constexpr size_t OFF_W2H    = OFF_W1H    + (size_t)C2 * C1 * 2;    // C3*C2 f16
constexpr size_t OFF_STATS  = OFF_W2H    + (size_t)C3 * C2 * 2;    // 1024 f32
constexpr size_t OFF_FEAT   = OFF_STATS  + 1024 * 4;               // M*C0P f16
constexpr size_t OFF_Y0     = OFF_FEAT   + (size_t)M * C0P * 2;    // M*C1  f16
constexpr size_t OFF_Y1     = OFF_Y0     + (size_t)M * C1 * 2;     // M*C2  f16
constexpr size_t OFF_Y2     = OFF_Y1     + (size_t)M * C2 * 2;     // M*C3  f16

constexpr size_t OUT1_OFF = (size_t)B * 3 * S;                     // floats

// ---------------- CDNA5 async copy helpers ----------------
// GVS mode: mem_addr = SGPR_base + VGPR_offset + inst_offset, dest = LDS.
// Tracked by ASYNCcnt; LDS byte address comes from the VDST VGPR.
DEV void async_ld_b128(unsigned lds_off, const void* gbase, unsigned gbyteoff) {
  asm volatile("global_load_async_to_lds_b128 %0, %1, %2"
               :: "v"(lds_off), "v"(gbyteoff), "s"(gbase)
               : "memory");
}
DEV void wait_async0() { asm volatile("s_wait_asynccnt 0" ::: "memory"); }

// lower 32 bits of a generic pointer into LDS == LDS byte offset
// (LDS aperture occupies addr[63:32] only; ISA §10.2 aperture mapping)
DEV unsigned lds_addr_of(const void* p) { return (unsigned)(uintptr_t)p; }

// ---------------- WMMA helpers ----------------
DEV v16h cat16(v8h lo, v8h hi) {
  v16h r;
#pragma unroll
  for (int i = 0; i < 8; ++i) { r[i] = lo[i]; r[i + 8] = hi[i]; }
  return r;
}

DEV v8f wmma_f16(v16h a, v16h b, v8f c) {
  // D = A(16x32 f16) * B(32x16 f16) + C(16x16 f32)
  return __builtin_amdgcn_wmma_f32_16x16x32_f16(
      false, a, false, b, (short)0, c, false, false);
}

// A operand: row-major activations [rows x ld], ISA 16-bit A layout:
// lane<16 holds M=lane, K = {c..c+7, c+16..c+23} with c = kt*32
// lane>=16 holds M=lane-16, K = {c+8..c+15, c+24..c+31}
DEV v16h load_a_raw(const _Float16* __restrict__ A, int ld, int m_base,
                    int kt, int lane) {
  int row = m_base + (lane & 15);
  int c   = kt * 32 + ((lane & 16) ? 8 : 0);
  const _Float16* p = A + (size_t)row * ld + c;
  v8h lo = *(const v8h*)p;
  v8h hi = *(const v8h*)(p + 16);
  return cat16(lo, hi);
}

// A operand with per-channel BN affine + ReLU applied on the fly
DEV v16h load_a_bn(const _Float16* __restrict__ Y, int ld, int m_base, int kt,
                   int lane, const float* __restrict__ sc,
                   const float* __restrict__ sh) {
  int row = m_base + (lane & 15);
  int c   = kt * 32 + ((lane & 16) ? 8 : 0);
  const _Float16* p = Y + (size_t)row * ld + c;
  v16h r;
#pragma unroll
  for (int i = 0; i < 8; ++i) {
    float v0 = (float)p[i]      * sc[c + i]      + sh[c + i];
    float v1 = (float)p[16 + i] * sc[c + 16 + i] + sh[c + 16 + i];
    r[i]     = (_Float16)fmaxf(v0, 0.f);
    r[i + 8] = (_Float16)fmaxf(v1, 0.f);
  }
  return r;
}

// B operand = W^T. W row-major [Co x ld]; lane<16 holds N=lane,
// K = {c..c+7, c+16..c+23}; lane>=16 holds K = {c+8..c+15, c+24..c+31}.
DEV v16h load_b(const _Float16* __restrict__ W, int ld, int n_base, int kt,
                int lane) {
  int o = n_base + (lane & 15);
  int c = kt * 32 + ((lane & 16) ? 8 : 0);
  const _Float16* p = W + (size_t)o * ld + c;
  v8h lo = *(const v8h*)p;
  v8h hi = *(const v8h*)(p + 16);
  return cat16(lo, hi);
}

// ---------------- kernel 0: init (zero stats, convert weights to f16) ----
__global__ void k_init(const float* __restrict__ W0, const float* __restrict__ W1,
                       const float* __restrict__ W2, _Float16* __restrict__ W0h,
                       _Float16* __restrict__ W1h, _Float16* __restrict__ W2h,
                       float* __restrict__ stats) {
  int i = blockIdx.x * blockDim.x + threadIdx.x;
  if (i < 1024) stats[i] = 0.f;
  if (i < C1 * C0P) {
    int o = i / C0P, c = i % C0P;
    // feat layout: cols 0..63 = point feats (orig channels 3..66),
    //              cols 64..66 = centered xyz (orig 0..2), 67..95 zero pad
    float v = 0.f;
    if (c < 64)      v = W0[o * C0 + (c + 3)];
    else if (c < 67) v = W0[o * C0 + (c - 64)];
    W0h[i] = (_Float16)v;
  }
  if (i < C2 * C1) W1h[i] = (_Float16)W1[i];
  if (i < C3 * C2) W2h[i] = (_Float16)W2[i];
}

// ---------------- kernel 1: transpose xyz / points ----------------
__global__ void k_transpose(const float* __restrict__ xyz,
                            const float* __restrict__ pts,
                            float* __restrict__ xyzT,
                            _Float16* __restrict__ ptsT) {
  int t = blockIdx.x * 256 + threadIdx.x;  // over B*N
  int n = t & (N - 1);
  int b = t >> 12;
#pragma unroll
  for (int c = 0; c < 3; ++c)
    xyzT[((size_t)b * N + n) * 3 + c] = xyz[((size_t)b * 3 + c) * N + n];
  for (int c = 0; c < D; ++c)
    ptsT[((size_t)b * N + n) * D + c] =
        (_Float16)pts[((size_t)b * D + c) * N + n];
}

// ---------------- kernel 2: farthest point sampling ----------------
// One workgroup per batch. Distances & coords live in registers
// (16 points / thread, strided). Argmax = shfl-xor wave reduce + LDS.
__global__ __launch_bounds__(256) void k_fps(const float* __restrict__ xyzT,
                                             float* __restrict__ new_xyz,
                                             float* __restrict__ outNew) {
  __shared__ float cpos[3];
  __shared__ float wval[8];
  __shared__ int   widx[8];
  __shared__ int   s_far;

  int b = blockIdx.x, tid = threadIdx.x;
  int lane = tid & 31, wave = tid >> 5;
  const float* base = xyzT + (size_t)b * N * 3;

  float px[16], py[16], pz[16], dist[16];
#pragma unroll
  for (int i = 0; i < 16; ++i) {
    int n = i * 256 + tid;
    px[i] = base[n * 3 + 0];
    py[i] = base[n * 3 + 1];
    pz[i] = base[n * 3 + 2];
    dist[i] = 1e10f;
  }

  int far = 0;
  for (int t = 0; t < S; ++t) {
    if (tid == 0) {
      float cx = base[far * 3 + 0], cy = base[far * 3 + 1],
            cz = base[far * 3 + 2];
      cpos[0] = cx; cpos[1] = cy; cpos[2] = cz;
      new_xyz[((size_t)b * S + t) * 3 + 0] = cx;
      new_xyz[((size_t)b * S + t) * 3 + 1] = cy;
      new_xyz[((size_t)b * S + t) * 3 + 2] = cz;
      outNew[((size_t)b * 3 + 0) * S + t] = cx;
      outNew[((size_t)b * 3 + 1) * S + t] = cy;
      outNew[((size_t)b * 3 + 2) * S + t] = cz;
    }
    __syncthreads();
    float cx = cpos[0], cy = cpos[1], cz = cpos[2];

    float bestv = -1.f;
    int   besti = 0x7fffffff;
#pragma unroll
    for (int i = 0; i < 16; ++i) {
      float dx = px[i] - cx, dy = py[i] - cy, dz = pz[i] - cz;
      float d = dx * dx + dy * dy + dz * dz;
      dist[i] = fminf(dist[i], d);
      if (dist[i] > bestv) { bestv = dist[i]; besti = i * 256 + tid; }
    }
    // wave32 reduce (argmax, first-index tie-break like jnp.argmax)
#pragma unroll
    for (int m = 16; m >= 1; m >>= 1) {
      float ov = __shfl_xor(bestv, m, 32);
      int   oi = __shfl_xor(besti, m, 32);
      if (ov > bestv || (ov == bestv && oi < besti)) { bestv = ov; besti = oi; }
    }
    if (lane == 0) { wval[wave] = bestv; widx[wave] = besti; }
    __syncthreads();
    if (tid == 0) {
      float bv = wval[0]; int bi = widx[0];
#pragma unroll
      for (int w = 1; w < 8; ++w)
        if (wval[w] > bv || (wval[w] == bv && widx[w] < bi)) {
          bv = wval[w]; bi = widx[w];
        }
      s_far = bi;
    }
    __syncthreads();
    far = s_far;
  }
}

// ---------------- kernel 3: ball query + grouping -> feat matrix ---------
DEV void write_feat_row(_Float16* __restrict__ dst,
                        const _Float16* __restrict__ ptsrow, float dx,
                        float dy, float dz) {
#pragma unroll
  for (int c = 0; c < 64; c += 8)
    *(v8h*)(dst + c) = *(const v8h*)(ptsrow + c);
  dst[64] = (_Float16)dx;
  dst[65] = (_Float16)dy;
  dst[66] = (_Float16)dz;
#pragma unroll
  for (int c = 67; c < 72; ++c) dst[c] = (_Float16)0.f;
  v8h z = {};
#pragma unroll
  for (int c = 72; c < 96; c += 8) *(v8h*)(dst + c) = z;
}

__global__ __launch_bounds__(256) void k_ballgroup(
    const float* __restrict__ xyzT, const _Float16* __restrict__ ptsT,
    const float* __restrict__ new_xyz, _Float16* __restrict__ feat) {
  int tid = threadIdx.x, lane = tid & 31;
  int wg = blockIdx.x * 8 + (tid >> 5);  // one wave per (b,s)
  int b = wg >> 10, s = wg & (S - 1);

  float cx = new_xyz[((size_t)b * S + s) * 3 + 0];
  float cy = new_xyz[((size_t)b * S + s) * 3 + 1];
  float cz = new_xyz[((size_t)b * S + s) * 3 + 2];

  const float*    xb = xyzT + (size_t)b * N * 3;
  const _Float16* pb = ptsT + (size_t)b * N * D;
  _Float16* frow = feat + (size_t)((size_t)b * S + s) * K * C0P;

  unsigned cnt = 0;
  for (int bs = 0; bs < N; bs += 32) {
    int n = bs + lane;
    float x = xb[n * 3 + 0], y = xb[n * 3 + 1], z = xb[n * 3 + 2];
    float dx = x - cx, dy = y - cy, dz = z - cz;
    float d2 = dx * dx + dy * dy + dz * dz;
    bool in = (d2 <= R2);
    unsigned mask = __builtin_amdgcn_ballot_w32(in);
    unsigned pos = cnt + __popc(mask & ((1u << lane) - 1u));
    if (in && pos < (unsigned)K)
      write_feat_row(frow + (size_t)pos * C0P, pb + (size_t)n * D, dx, dy, dz);
    cnt += __popc(mask);
    if (cnt >= (unsigned)K) break;
  }
  if (cnt < (unsigned)K) {
    // pad with point N-1 (reference's sorted-sentinel semantics)
    int n = N - 1;
    float dx = xb[n * 3 + 0] - cx, dy = xb[n * 3 + 1] - cy,
          dz = xb[n * 3 + 2] - cz;
    for (unsigned slot = cnt + lane; slot < (unsigned)K; slot += 32)
      write_feat_row(frow + (size_t)slot * C0P, pb + (size_t)n * D, dx, dy, dz);
  }
}

// ---------------- GEMM epilogue: store f16 + per-channel partial stats ----
DEV void epilogue(v8f acc, _Float16* __restrict__ y, int ld, int m_base,
                  int n_base, int lane, float* lsum, float* lsq) {
  int o  = n_base + (lane & 15);
  int mo = m_base + ((lane & 16) ? 8 : 0);
  float s = 0.f, q = 0.f;
#pragma unroll
  for (int j = 0; j < 8; ++j) {
    float v = acc[j];
    y[(size_t)(mo + j) * ld + o] = (_Float16)v;
    s += v;
    q += v * v;
  }
  atomicAdd(&lsum[o], s);
  atomicAdd(&lsq[o], q);
}

// ---------------- kernel 4: GEMM layer 0 (feat[96] -> y0[64]) ----------
// A tile (32 x 96 f16, 6 KB) staged once per block into LDS via the CDNA5
// async copy engine; 2 M-waves x 4 N-waves consume it from LDS.
__global__ __launch_bounds__(256) void k_gemm0(
    const _Float16* __restrict__ feat, const _Float16* __restrict__ Wh,
    _Float16* __restrict__ y, float* __restrict__ gsum,
    float* __restrict__ gsq) {
  __shared__ alignas(16) _Float16 At[32 * C0P];
  __shared__ float lsum[C1], lsq[C1];
  int tid = threadIdx.x, lane = tid & 31, wave = tid >> 5;
  if (tid < C1) { lsum[tid] = 0.f; lsq[tid] = 0.f; }

  int m0 = blockIdx.x * 32;
  unsigned lds0 = lds_addr_of(At);
  unsigned gb   = (unsigned)(m0 * (C0P * 2));
#pragma unroll
  for (int seg = tid; seg < (32 * C0P * 2) / 16; seg += 256)
    async_ld_b128(lds0 + seg * 16, feat, gb + seg * 16);
  wait_async0();
  __syncthreads();

  int mloc   = (wave >> 2) * 16;
  int n_base = (wave & 3) * 16;
  __builtin_prefetch(Wh + (size_t)(n_base + (lane & 15)) * C0P, 0, 3);

  v8f acc = {};
#pragma unroll
  for (int kt = 0; kt < 3; ++kt) {
    v16h a  = load_a_raw(At, C0P, mloc, kt, lane);   // ds_load_b128
    v16h bm = load_b(Wh, C0P, n_base, kt, lane);
    acc = wmma_f16(a, bm, acc);
  }
  epilogue(acc, y, C1, m0 + mloc, n_base, lane, lsum, lsq);
  __syncthreads();
  if (tid < C1) { atomicAdd(&gsum[tid], lsum[tid]); atomicAdd(&gsq[tid], lsq[tid]); }
}

// ---------------- kernel 5: GEMM layer 1 (BN0+ReLU(y0)[64] -> y1[64]) ----
__global__ __launch_bounds__(256) void k_gemm1(
    const _Float16* __restrict__ y0, const _Float16* __restrict__ Wh,
    _Float16* __restrict__ y1, const float* __restrict__ sc,
    const float* __restrict__ sh, float* __restrict__ gsum,
    float* __restrict__ gsq) {
  __shared__ alignas(16) _Float16 At[32 * C1];
  __shared__ float lsum[C2], lsq[C2], ssc[C1], ssh[C1];
  int tid = threadIdx.x, lane = tid & 31, wave = tid >> 5;
  if (tid < C2) { lsum[tid] = 0.f; lsq[tid] = 0.f; }
  if (tid < C1) { ssc[tid] = sc[tid]; ssh[tid] = sh[tid]; }

  int m0 = blockIdx.x * 32;
  // stage raw (pre-BN) A tile: 32 x 64 f16 = 4 KB = 256 x 16B segments
  async_ld_b128(lds_addr_of(At) + tid * 16, y0,
                (unsigned)(m0 * (C1 * 2)) + tid * 16);
  wait_async0();
  __syncthreads();

  int mloc   = (wave >> 2) * 16;
  int n_base = (wave & 3) * 16;
  __builtin_prefetch(Wh + (size_t)(n_base + (lane & 15)) * C1, 0, 3);

  v8f acc = {};
#pragma unroll
  for (int kt = 0; kt < 2; ++kt) {
    v16h a  = load_a_bn(At, C1, mloc, kt, lane, ssc, ssh);
    v16h bm = load_b(Wh, C1, n_base, kt, lane);
    acc = wmma_f16(a, bm, acc);
  }
  epilogue(acc, y1, C2, m0 + mloc, n_base, lane, lsum, lsq);
  __syncthreads();
  if (tid < C2) { atomicAdd(&gsum[tid], lsum[tid]); atomicAdd(&gsq[tid], lsq[tid]); }
}

// ---------------- kernel 6: GEMM layer 2 (BN1+ReLU(y1)[64] -> y2[128]) ---
__global__ __launch_bounds__(256) void k_gemm2(
    const _Float16* __restrict__ y1, const _Float16* __restrict__ Wh,
    _Float16* __restrict__ y2, const float* __restrict__ sc,
    const float* __restrict__ sh, float* __restrict__ gsum,
    float* __restrict__ gsq) {
  __shared__ alignas(16) _Float16 At[16 * C2];
  __shared__ float lsum[C3], lsq[C3], ssc[C2], ssh[C2];
  int tid = threadIdx.x, lane = tid & 31, wave = tid >> 5;
  if (tid < C3) { lsum[tid] = 0.f; lsq[tid] = 0.f; }
  if (tid < C2) { ssc[tid] = sc[tid]; ssh[tid] = sh[tid]; }

  int m0 = blockIdx.x * 16;  // 16 rows shared by all 8 waves
  // stage A tile: 16 x 64 f16 = 2 KB = 128 x 16B segments
  if (tid < 128)
    async_ld_b128(lds_addr_of(At) + tid * 16, y1,
                  (unsigned)(m0 * (C2 * 2)) + tid * 16);
  wait_async0();
  __syncthreads();

  int n_base = wave * 16;  // 128 output channels
  __builtin_prefetch(Wh + (size_t)(n_base + (lane & 15)) * C2, 0, 3);

  v8f acc = {};
#pragma unroll
  for (int kt = 0; kt < 2; ++kt) {
    v16h a  = load_a_bn(At, C2, 0, kt, lane, ssc, ssh);
    v16h bm = load_b(Wh, C2, n_base, kt, lane);
    acc = wmma_f16(a, bm, acc);
  }
  epilogue(acc, y2, C3, m0, n_base, lane, lsum, lsq);
  __syncthreads();
  if (tid < C3) { atomicAdd(&gsum[tid], lsum[tid]); atomicAdd(&gsq[tid], lsq[tid]); }
}

// ---------------- kernel 7: finalize BN stats -> scale/shift ------------
__global__ void k_finalize(const float* __restrict__ sum,
                           const float* __restrict__ sq,
                           const float* __restrict__ g,
                           const float* __restrict__ be,
                           float* __restrict__ scale,
                           float* __restrict__ shift, int C) {
  int i = threadIdx.x;
  if (i < C) {
    float inv = 1.f / (float)M;
    float mean = sum[i] * inv;
    float var  = sq[i] * inv - mean * mean;   // population var == jnp.var
    float sc   = g[i] * rsqrtf(var + EPSV);
    scale[i] = sc;
    shift[i] = be[i] - mean * sc;
  }
}

// ---------------- kernel 8: BN2 + ReLU + max-pool over K ----------------
__global__ void k_maxpool(const _Float16* __restrict__ y2,
                          const float* __restrict__ sc,
                          const float* __restrict__ sh,
                          float* __restrict__ out) {
  int t = blockIdx.x * 256 + threadIdx.x;   // over B*S*C3
  int o = t & (C3 - 1);
  int s = (t >> 7) & (S - 1);
  int b = t >> 17;
  const _Float16* p = y2 + (size_t)((size_t)(b * S + s) * K) * C3 + o;
  float scv = sc[o], shv = sh[o];
  float m = 0.f;  // relu outputs are >= 0
#pragma unroll
  for (int k = 0; k < K; ++k) {
    float v = (float)p[(size_t)k * C3] * scv + shv;
    m = fmaxf(m, fmaxf(v, 0.f));
  }
  out[OUT1_OFF + ((size_t)(b * C3 + o)) * S + s] = m;
}

// ---------------- host driver ----------------
extern "C" void kernel_launch(void* const* d_in, const int* in_sizes, int n_in,
                              void* d_out, int out_size, void* d_ws,
                              size_t ws_size, hipStream_t stream) {
  const float* xyz = (const float*)d_in[0];
  const float* pts = (const float*)d_in[1];
  const float* W0  = (const float*)d_in[2];
  const float* g0  = (const float*)d_in[4];
  const float* be0 = (const float*)d_in[5];
  const float* W1  = (const float*)d_in[6];
  const float* g1  = (const float*)d_in[8];
  const float* be1 = (const float*)d_in[9];
  const float* W2  = (const float*)d_in[10];
  const float* g2  = (const float*)d_in[12];
  const float* be2 = (const float*)d_in[13];
  float* out = (float*)d_out;

  char* w = (char*)d_ws;
  float*    new_xyz = (float*)(w + OFF_NEWXYZ);
  float*    xyzT    = (float*)(w + OFF_XYZT);
  _Float16* ptsT    = (_Float16*)(w + OFF_PTST);
  _Float16* W0h     = (_Float16*)(w + OFF_W0H);
  _Float16* W1h     = (_Float16*)(w + OFF_W1H);
  _Float16* W2h     = (_Float16*)(w + OFF_W2H);
  float*    st      = (float*)(w + OFF_STATS);
  _Float16* feat    = (_Float16*)(w + OFF_FEAT);
  _Float16* y0      = (_Float16*)(w + OFF_Y0);
  _Float16* y1      = (_Float16*)(w + OFF_Y1);
  _Float16* y2      = (_Float16*)(w + OFF_Y2);

  float *sum0 = st,        *sq0 = st + 64,  *sc0 = st + 128, *sh0 = st + 192;
  float *sum1 = st + 256,  *sq1 = st + 320, *sc1 = st + 384, *sh1 = st + 448;
  float *sum2 = st + 512,  *sq2 = st + 640, *sc2 = st + 768, *sh2 = st + 896;

  k_init<<<32, 256, 0, stream>>>(W0, W1, W2, W0h, W1h, W2h, st);
  k_transpose<<<(B * N) / 256, 256, 0, stream>>>(xyz, pts, xyzT, ptsT);
  k_fps<<<B, 256, 0, stream>>>(xyzT, new_xyz, out);
  k_ballgroup<<<(B * S) / 8, 256, 0, stream>>>(xyzT, ptsT, new_xyz, feat);

  k_gemm0<<<M / 32, 256, 0, stream>>>(feat, W0h, y0, sum0, sq0);
  k_finalize<<<1, 128, 0, stream>>>(sum0, sq0, g0, be0, sc0, sh0, C1);

  k_gemm1<<<M / 32, 256, 0, stream>>>(y0, W1h, y1, sc0, sh0, sum1, sq1);
  k_finalize<<<1, 128, 0, stream>>>(sum1, sq1, g1, be1, sc1, sh1, C2);

  k_gemm2<<<M / 16, 256, 0, stream>>>(y1, W2h, y2, sc1, sh1, sum2, sq2);
  k_finalize<<<1, 128, 0, stream>>>(sum2, sq2, g2, be2, sc2, sh2, C3);

  k_maxpool<<<(B * S * C3) / 256, 256, 0, stream>>>(y2, sc2, sh2, out);
}
